// InceptionBlock_5463198400893
// MI455X (gfx1250) — matrix operand order repaired and checked by reading, not compiled, not run on hardware
//
#include <hip/hip_runtime.h>

#define NN 8192   // nodes
#define FF 128    // features (F_IN == FILTERS)

typedef __attribute__((ext_vector_type(16))) __bf16 v16bf;
typedef __attribute__((ext_vector_type(8)))  __bf16 v8bf;
typedef __attribute__((ext_vector_type(8)))  float  v8f;
typedef __attribute__((ext_vector_type(4)))  float  v4f;

static __device__ __forceinline__ __bf16 f2bf(float f) {
  unsigned u = __builtin_bit_cast(unsigned, f);
  u += 0x7fffu + ((u >> 16) & 1u);               // round-to-nearest-even
  unsigned short s = (unsigned short)(u >> 16);
  return __builtin_bit_cast(__bf16, s);
}
static __device__ __forceinline__ float bf2f(__bf16 b) {
  unsigned u = (unsigned)__builtin_bit_cast(unsigned short, b) << 16;
  return __builtin_bit_cast(float, u);
}
// v ~= hi + lo  (two bf16s capture ~16 mantissa bits of an fp32)
struct bfpair { __bf16 hi, lo; };
static __device__ __forceinline__ bfpair split2(float v) {
  bfpair r;
  r.hi = f2bf(v);
  r.lo = f2bf(v - bf2f(r.hi));
  return r;
}

// ---- prep: A fp32 -> bf16 (exact: entries are 0.0/1.0), 8 elems/thread ----
__global__ __launch_bounds__(256) void k_convA(const float* __restrict__ A,
                                               __bf16* __restrict__ Ab) {
  size_t i = (size_t)blockIdx.x * 256 + threadIdx.x;   // i indexes groups of 8
  const v4f* p = (const v4f*)A + i * 2;
  v4f a = p[0], b = p[1];
  v8bf o;
  o[0]=f2bf(a[0]); o[1]=f2bf(a[1]); o[2]=f2bf(a[2]); o[3]=f2bf(a[3]);
  o[4]=f2bf(b[0]); o[5]=f2bf(b[1]); o[6]=f2bf(b[2]); o[7]=f2bf(b[3]);
  ((v8bf*)Ab)[i] = o;
}

// ---- prep: x [NN][FF] fp32 -> xT hi/lo [FF][NN] bf16 ----
__global__ __launch_bounds__(256) void k_convX(const float* __restrict__ x,
                                               __bf16* __restrict__ xThi,
                                               __bf16* __restrict__ xTlo) {
  int i = blockIdx.x * 256 + threadIdx.x;   // < NN*FF
  int m = i >> 7, f = i & 127;
  bfpair s = split2(x[i]);
  xThi[(size_t)f * NN + m] = s.hi;
  xTlo[(size_t)f * NN + m] = s.lo;
}

// ---- prep: W1/W2/W3 [j][f] fp32 -> Wt hi/lo [3][f][j] bf16 (transposed) ----
__global__ __launch_bounds__(256) void k_convW(const float* __restrict__ W1,
                                               const float* __restrict__ W2,
                                               const float* __restrict__ W3,
                                               __bf16* __restrict__ Wthi,
                                               __bf16* __restrict__ Wtlo) {
  int i = blockIdx.x * 256 + threadIdx.x;   // < 3*FF*FF
  int b = i >> 14;
  int rem = i & 16383;
  int j = rem >> 7, f = rem & 127;
  const float* W = (b == 0) ? W1 : ((b == 1) ? W2 : W3);
  bfpair s = split2(W[j * FF + f]);
  Wthi[(size_t)b * FF * FF + (size_t)f * FF + j] = s.hi;
  Wtlo[(size_t)b * FF * FF + (size_t)f * FF + j] = s.lo;
}

// ---- propagation step: y_next = A @ y_prev (A exact bf16, y as hi+lo split) -
// yThi/yTlo: [FF][NN] bf16 (K-contiguous for B fragments)
// outputs: transposed hi/lo (for next step) and normal-layout hi/lo (for final)
// one wave computes a 32(M) x 128(F) strip; 256 waves cover M=8192.
template <int A_IS_BF16>
__global__ __launch_bounds__(256)
void k_prop(const void* __restrict__ Aany,
            const __bf16* __restrict__ yThi, const __bf16* __restrict__ yTlo,
            __bf16* __restrict__ oThi, __bf16* __restrict__ oTlo,
            __bf16* __restrict__ oNhi, __bf16* __restrict__ oNlo) {
  const int wave = blockIdx.x * 8 + (threadIdx.x >> 5);   // 0..255
  const int lane = threadIdx.x & 31;
  const int r = lane & 15;
  const int half = lane >> 4;
  const int m0 = wave * 32;

  v8f acc[2][8];
  #pragma unroll
  for (int i = 0; i < 2; ++i)
    #pragma unroll
    for (int t = 0; t < 8; ++t) acc[i][t] = (v8f){};

  for (int k0 = 0; k0 < NN; k0 += 32) {
    // two A fragments: rows [m0, m0+16) and [m0+16, m0+32), cols [k0, k0+32)
    v16bf afrag[2];
    #pragma unroll
    for (int i = 0; i < 2; ++i) {
      const int arow = m0 + i * 16 + r;
      if (A_IS_BF16) {
        const __bf16* ap = (const __bf16*)Aany + (size_t)arow * NN + k0 + half * 16;
        v8bf lo = *(const v8bf*)ap;          // K = base+0..7
        v8bf hi = *(const v8bf*)(ap + 8);    // K = base+8..15
        #pragma unroll
        for (int e = 0; e < 8; ++e) { afrag[i][e] = lo[e]; afrag[i][8 + e] = hi[e]; }
      } else {
        const float* ap = (const float*)Aany + (size_t)arow * NN + k0 + half * 16;
        v4f l0 = *(const v4f*)ap,        l1 = *(const v4f*)(ap + 4);
        v4f h0 = *(const v4f*)(ap + 8),  h1 = *(const v4f*)(ap + 12);
        #pragma unroll
        for (int e = 0; e < 4; ++e) {
          afrag[i][e]      = f2bf(l0[e]);  afrag[i][4 + e]  = f2bf(l1[e]);
          afrag[i][8 + e]  = f2bf(h0[e]);  afrag[i][12 + e] = f2bf(h1[e]);
        }
      }
    }
    // 8 F-tiles: B hi/lo fragments shared by both M-tiles
    #pragma unroll
    for (int t = 0; t < 8; ++t) {
      const size_t boff = (size_t)(t * 16 + r) * NN + k0 + half * 16;
      v16bf bhi = *(const v16bf*)(yThi + boff);   // 16 contiguous K values
      v16bf blo = *(const v16bf*)(yTlo + boff);
      acc[0][t] = __builtin_amdgcn_wmma_f32_16x16x32_bf16(
          false, afrag[0], false, bhi, (short)0, acc[0][t], false, false);
      acc[1][t] = __builtin_amdgcn_wmma_f32_16x16x32_bf16(
          false, afrag[1], false, bhi, (short)0, acc[1][t], false, false);
      acc[0][t] = __builtin_amdgcn_wmma_f32_16x16x32_bf16(
          false, afrag[0], false, blo, (short)0, acc[0][t], false, false);
      acc[1][t] = __builtin_amdgcn_wmma_f32_16x16x32_bf16(
          false, afrag[1], false, blo, (short)0, acc[1][t], false, false);
    }
  }

  #pragma unroll
  for (int i = 0; i < 2; ++i) {
    #pragma unroll
    for (int t = 0; t < 8; ++t) {
      const int f = t * 16 + r;
      const int mb = m0 + i * 16 + half * 8;
      v8bf ohi, olo;
      #pragma unroll
      for (int v = 0; v < 8; ++v) {
        bfpair s = split2(acc[i][t][v]);
        ohi[v] = s.hi;
        olo[v] = s.lo;
      }
      // transposed stores: 8 consecutive M at row f -> one 16B store each
      *(v8bf*)(oThi + (size_t)f * NN + mb) = ohi;
      *(v8bf*)(oTlo + (size_t)f * NN + mb) = olo;
      // normal-layout stores for the final GEMM (strided b16)
      #pragma unroll
      for (int v = 0; v < 8; ++v) {
        oNhi[(size_t)(mb + v) * FF + f] = ohi[v];
        oNlo[(size_t)(mb + v) * FF + f] = olo[v];
      }
    }
  }
}

// ---- final: out = (y1@W1 + y2@W2 + y3@W3) / 3, fp32 -------------------------
// split product: y@W ~= yhi@Whi + ylo@Whi + yhi@Wlo
__global__ __launch_bounds__(256)
void k_final(const __bf16* __restrict__ ynhi, const __bf16* __restrict__ ynlo,
             const __bf16* __restrict__ Wthi, const __bf16* __restrict__ Wtlo,
             float* __restrict__ out) {
  const int wave = blockIdx.x * 8 + (threadIdx.x >> 5);   // 0..511
  const int lane = threadIdx.x & 31;
  const int r = lane & 15;
  const int half = lane >> 4;
  const int m0 = wave * 16;

  v8f acc[8];
  #pragma unroll
  for (int t = 0; t < 8; ++t) acc[t] = (v8f){};

  for (int b = 0; b < 3; ++b) {
    const __bf16* yhi = ynhi + (size_t)b * NN * FF;
    const __bf16* ylo = ynlo + (size_t)b * NN * FF;
    const __bf16* whi = Wthi + (size_t)b * FF * FF;
    const __bf16* wlo = Wtlo + (size_t)b * FF * FF;
    #pragma unroll
    for (int k0 = 0; k0 < FF; k0 += 32) {
      const size_t aoff = (size_t)(m0 + r) * FF + k0 + half * 16;
      v16bf ahi, alo;
      {
        v8bf l = *(const v8bf*)(yhi + aoff);
        v8bf h = *(const v8bf*)(yhi + aoff + 8);
        #pragma unroll
        for (int e = 0; e < 8; ++e) { ahi[e] = l[e]; ahi[8 + e] = h[e]; }
      }
      {
        v8bf l = *(const v8bf*)(ylo + aoff);
        v8bf h = *(const v8bf*)(ylo + aoff + 8);
        #pragma unroll
        for (int e = 0; e < 8; ++e) { alo[e] = l[e]; alo[8 + e] = h[e]; }
      }
      #pragma unroll
      for (int t = 0; t < 8; ++t) {
        const size_t boff = (size_t)(t * 16 + r) * FF + k0 + half * 16;
        v16bf bhi = *(const v16bf*)(whi + boff);
        v16bf blo = *(const v16bf*)(wlo + boff);
        acc[t] = __builtin_amdgcn_wmma_f32_16x16x32_bf16(
            false, ahi, false, bhi, (short)0, acc[t], false, false);
        acc[t] = __builtin_amdgcn_wmma_f32_16x16x32_bf16(
            false, alo, false, bhi, (short)0, acc[t], false, false);
        acc[t] = __builtin_amdgcn_wmma_f32_16x16x32_bf16(
            false, ahi, false, blo, (short)0, acc[t], false, false);
      }
    }
  }

  #pragma unroll
  for (int t = 0; t < 8; ++t) {
    const int f = t * 16 + r;
    #pragma unroll
    for (int v = 0; v < 8; ++v)
      out[(size_t)(m0 + half * 8 + v) * FF + f] = acc[t][v] * (1.0f / 3.0f);
  }
}

extern "C" void kernel_launch(void* const* d_in, const int* in_sizes, int n_in,
                              void* d_out, int out_size, void* d_ws, size_t ws_size,
                              hipStream_t stream) {
  (void)in_sizes; (void)n_in; (void)out_size;
  const float* x  = (const float*)d_in[0];
  const float* A  = (const float*)d_in[1];
  const float* W1 = (const float*)d_in[2];
  const float* W2 = (const float*)d_in[3];
  const float* W3 = (const float*)d_in[4];
  float* out = (float*)d_out;

  char* p = (char*)d_ws;
  const size_t ySz = (size_t)NN * FF * sizeof(__bf16);   // 2 MiB each
  __bf16* xThi = (__bf16*)p; p += ySz;
  __bf16* xTlo = (__bf16*)p; p += ySz;
  // transposed hi/lo ping-pong buffers for the 3 hops
  __bf16* yT[3][2];
  for (int s = 0; s < 3; ++s)
    for (int h = 0; h < 2; ++h) { yT[s][h] = (__bf16*)p; p += ySz; }
  // normal-layout hi/lo, concatenated [3][NN][FF] for k_final
  __bf16* ynhi = (__bf16*)p; p += 3 * ySz;
  __bf16* ynlo = (__bf16*)p; p += 3 * ySz;
  __bf16* Wthi = (__bf16*)p; p += (size_t)3 * FF * FF * sizeof(__bf16);
  __bf16* Wtlo = (__bf16*)p; p += (size_t)3 * FF * FF * sizeof(__bf16);
  const size_t baseBytes = (size_t)(p - (char*)d_ws);
  const size_t abBytes = (size_t)NN * NN * sizeof(__bf16);   // 128 MiB
  __bf16* Ab = (__bf16*)p;
  const bool useAb = (ws_size >= baseBytes + abBytes);

  k_convX<<<(NN * FF) / 256, 256, 0, stream>>>(x, xThi, xTlo);
  k_convW<<<(3 * FF * FF) / 256, 256, 0, stream>>>(W1, W2, W3, Wthi, Wtlo);

  if (useAb) {
    // bf16 A (128 MiB) is L2-resident on MI455X (192 MiB L2): steps 2+3 hit L2
    k_convA<<<(int)(((size_t)NN * NN / 8) / 256), 256, 0, stream>>>(A, Ab);
    k_prop<1><<<32, 256, 0, stream>>>((const void*)Ab, xThi, xTlo,
                                      yT[0][0], yT[0][1], ynhi + 0 * NN * FF, ynlo + 0 * NN * FF);
    k_prop<1><<<32, 256, 0, stream>>>((const void*)Ab, yT[0][0], yT[0][1],
                                      yT[1][0], yT[1][1], ynhi + 1 * NN * FF, ynlo + 1 * NN * FF);
    k_prop<1><<<32, 256, 0, stream>>>((const void*)Ab, yT[1][0], yT[1][1],
                                      yT[2][0], yT[2][1], ynhi + 2 * NN * FF, ynlo + 2 * NN * FF);
  } else {
    // fallback: stream fp32 A and convert in-register
    k_prop<0><<<32, 256, 0, stream>>>((const void*)A, xThi, xTlo,
                                      yT[0][0], yT[0][1], ynhi + 0 * NN * FF, ynlo + 0 * NN * FF);
    k_prop<0><<<32, 256, 0, stream>>>((const void*)A, yT[0][0], yT[0][1],
                                      yT[1][0], yT[1][1], ynhi + 1 * NN * FF, ynlo + 1 * NN * FF);
    k_prop<0><<<32, 256, 0, stream>>>((const void*)A, yT[1][0], yT[1][1],
                                      yT[2][0], yT[2][1], ynhi + 2 * NN * FF, ynlo + 2 * NN * FF);
  }

  k_final<<<64, 256, 0, stream>>>(ynhi, ynlo, Wthi, Wtlo, out);
}
